// MPN_28561532518942
// MI455X (gfx1250) — matrix operand order repaired and checked by reading, not compiled, not run on hardware
//
#include <hip/hip_runtime.h>

// ---------------------------------------------------------------------------
// MPN on gfx1250: per-wave 16-edge tiles through two MLPs using
// v_wmma_f32_16x16x32_f16, weights pre-swizzled into B-fragment layout in LDS,
// activations staged per-wave in LDS in A-fragment-friendly layout, and the
// segment-sum done with global_atomic_add_f32 directly from the C fragments.
// ---------------------------------------------------------------------------

typedef __attribute__((ext_vector_type(16))) _Float16 v16h;
typedef __attribute__((ext_vector_type(8)))  float    v8f;

union PackH { _Float16 h[2]; unsigned int u; };
union AFrag { v16h v; unsigned int u[8]; };

#define WAVES   4      // waves per block (128 threads)
#define AE_S    98     // feat_e tile row stride (halves), odd dword count -> no bank conflicts
#define AN_S    66     // feat_n tile row stride
#define AH1_S   34     // h1 tile row stride
#define AH2_S   66     // h2 tile row stride
#define NFRAG   19     // total pre-swizzled B fragments

// --- A fragment (16x32 f16, MxK): lane = {half=lane>>4, m=lane&15},
//     element j=2v+lo: K = k0 + half*8 + (v>=4?16:0) + 2*(v&3) + lo  ---------
__device__ __forceinline__ v16h ldsAFrag(const _Float16* tile, int stride,
                                         int lane, int k0) {
  int half = (lane >> 4) & 1;
  int m = lane & 15;
  const _Float16* rp = tile + m * stride + k0 + half * 8;
  AFrag f;
#pragma unroll
  for (int v = 0; v < 8; ++v) {
    int k = ((v & 4) ? 16 : 0) + 2 * (v & 3);
    f.u[v] = *(const unsigned int*)(rp + k);   // even offset -> dword aligned
  }
  return f.v;
}

// --- B fragment, pre-swizzled in LDS as [frag][v][lane] dwords --------------
__device__ __forceinline__ v16h ldsBFrag(const unsigned int* wsw, int fidx,
                                         int lane) {
  AFrag f;
#pragma unroll
  for (int v = 0; v < 8; ++v) f.u[v] = wsw[fidx * 256 + v * 32 + lane];
  return f.v;
}

__device__ __forceinline__ v8f wmma16(v16h a, v16h b, v8f c) {
  return __builtin_amdgcn_wmma_f32_16x16x32_f16(false, a, false, b,
                                                (short)0, c, false, false);
}

__device__ __forceinline__ void st2(_Float16* p, float x, float y) {
  PackH pk; pk.h[0] = (_Float16)x; pk.h[1] = (_Float16)y;
  *(unsigned int*)p = pk.u;                    // even offsets only
}

// B-matrix (32x16 f16, KxN) fragment: lane = {half=lane>>4, n=lane&15},
// element j=2v+lo: K = ks + half*8 + (v>=4?16:0) + 2*(v&3) + lo.  Zero-pad outside [Krows,Ncols].
// 128 threads: t>>5 picks v-half, each thread fills v and v+4.
__device__ __forceinline__ void swzW(const float* __restrict__ W, int Krows,
                                     int Ncols, int ks, int n0,
                                     unsigned int* frag, int t) {
  int lane = t & 31;
  int vh = t >> 5;                 // 0..3
  int half = (lane >> 4) & 1;
  int n = n0 + (lane & 15);
#pragma unroll
  for (int i = 0; i < 2; ++i) {
    int v = vh + i * 4;
    int kb = ks + half * 8 + ((v & 4) ? 16 : 0) + 2 * (v & 3);
    float f0 = (kb     < Krows && n < Ncols) ? W[(size_t)kb * Ncols + n] : 0.f;
    float f1 = (kb + 1 < Krows && n < Ncols) ? W[(size_t)(kb + 1) * Ncols + n] : 0.f;
    PackH p; p.h[0] = (_Float16)f0; p.h[1] = (_Float16)f1;
    frag[v * 32 + lane] = p.u;
  }
}

__global__ __launch_bounds__(128) void mpn_wmma_kernel(
    const float* __restrict__ x_node, const float* __restrict__ x_edge,
    const int* __restrict__ src, const int* __restrict__ dst,
    const int* __restrict__ b_mask,
    const float* __restrict__ Wn1, const float* __restrict__ bn1,
    const float* __restrict__ Wn2, const float* __restrict__ bn2,
    const float* __restrict__ We1, const float* __restrict__ be1,
    const float* __restrict__ We2, const float* __restrict__ be2,
    float* __restrict__ out_nm, float* __restrict__ out_em,
    int E, int numTiles)
{
  __shared__ alignas(16) _Float16 Ae[WAVES][16 * AE_S];   // feat_e 16x96 (+pad)
  __shared__ alignas(16) _Float16 An[WAVES][16 * AN_S];   // feat_n 16x64
  __shared__ alignas(16) _Float16 Ah1[WAVES][16 * AH1_S]; // h1 16x32
  __shared__ alignas(16) _Float16 Ah2[WAVES][16 * AH2_S]; // h2 16x64
  __shared__ int dstL[WAVES][16];
  __shared__ unsigned int wsw[NFRAG * 256];               // swizzled weights

  const int t = threadIdx.x;
  const int lane = t & 31;
  const int w = t >> 5;

  // ---- pre-swizzle all weight fragments once per block ----
  // We1 (70x32, K padded to 96): frags 0..5 = (ki,ti)
#pragma unroll
  for (int ki = 0; ki < 3; ++ki)
#pragma unroll
    for (int ti = 0; ti < 2; ++ti)
      swzW(We1, 70, 32, ki * 32, ti * 16, &wsw[(ki * 2 + ti) * 256], t);
  // We2 (32x6, N padded to 16): frag 6
  swzW(We2, 32, 6, 0, 0, &wsw[6 * 256], t);
  // Wn1 (38x64, K padded to 64): frags 7..14
#pragma unroll
  for (int ki = 0; ki < 2; ++ki)
#pragma unroll
    for (int ti = 0; ti < 4; ++ti)
      swzW(Wn1, 38, 64, ki * 32, ti * 16, &wsw[(7 + ki * 4 + ti) * 256], t);
  // Wn2 (64x32): frags 15..18
#pragma unroll
  for (int ki = 0; ki < 2; ++ki)
#pragma unroll
    for (int ti = 0; ti < 2; ++ti)
      swzW(Wn2, 64, 32, ki * 32, ti * 16, &wsw[(15 + ki * 2 + ti) * 256], t);
  __syncthreads();

  // per-lane bias registers (C layout: lane&15 == output column)
  const int nl = lane & 15;
  const int half = (lane >> 4) & 1;
  const float bE1_0 = be1[nl], bE1_1 = be1[nl + 16];
  const float bE2   = (nl < 6) ? be2[nl] : 0.f;
  const float bN1_0 = bn1[nl], bN1_1 = bn1[nl + 16];
  const float bN1_2 = bn1[nl + 32], bN1_3 = bn1[nl + 48];
  const float bN2_0 = bn2[nl], bN2_1 = bn2[nl + 16];

  for (int tile = blockIdx.x * WAVES + w; tile < numTiles;
       tile += gridDim.x * WAVES) {
    const int base = tile * 16;

    // ---- gather: 2 lanes per edge row, float4 node-row loads ----
    {
      int r = lane >> 1, part = lane & 1;
      int e = base + r;
      bool valid = e < E;
      int ec = valid ? e : 0;
      int s = src[ec], d = dst[ec];
      bool back = b_mask[ec] != 0;
      const float4* ap = (const float4*)(x_node + (size_t)(back ? d : s) * 32) + part * 4;
      const float4* bp = (const float4*)(x_node + (size_t)(back ? s : d) * 32) + part * 4;
      _Float16* aeR = &Ae[w][r * AE_S];
      _Float16* anR = &An[w][r * AN_S];
#pragma unroll
      for (int q = 0; q < 4; ++q) {
        float4 av = ap[q], bv = bp[q];
        if (!valid) { av = float4{0.f,0.f,0.f,0.f}; bv = float4{0.f,0.f,0.f,0.f}; }
        int c0 = part * 16 + q * 4;
        st2(aeR + c0,          av.x, av.y); st2(aeR + c0 + 2,      av.z, av.w);
        st2(aeR + 32 + c0,     bv.x, bv.y); st2(aeR + 34 + c0,     bv.z, bv.w);
        float4 xdv = back ? av : bv;                 // xd == (back ? a : b)
        st2(anR + c0,          xdv.x, xdv.y); st2(anR + c0 + 2,    xdv.z, xdv.w);
      }
      if (part == 0) {
        const float* ep = x_edge + (size_t)ec * 6;
#pragma unroll
        for (int j = 0; j < 6; ++j) aeR[64 + j] = valid ? (_Float16)ep[j] : (_Float16)0.f;
#pragma unroll
        for (int j = 6; j < 16; ++j) aeR[64 + j] = (_Float16)0.f;
#pragma unroll
        for (int j = 0; j < 16; ++j) anR[32 + j] = (_Float16)0.f;  // em slot + pad
        dstL[w][r] = d;
      } else {
#pragma unroll
        for (int j = 0; j < 16; ++j) aeR[80 + j] = (_Float16)0.f;
#pragma unroll
        for (int j = 0; j < 16; ++j) anR[48 + j] = (_Float16)0.f;
      }
    }

    // ---- stage 1: h1 = relu(feat_e @ We1 + be1)  [16x32], 6 WMMA ----
    v8f c0 = {}, c1 = {};
#pragma unroll
    for (int ki = 0; ki < 3; ++ki) {
      v16h a = ldsAFrag(&Ae[w][0], AE_S, lane, ki * 32);
      c0 = wmma16(a, ldsBFrag(wsw, ki * 2 + 0, lane), c0);
      c1 = wmma16(a, ldsBFrag(wsw, ki * 2 + 1, lane), c1);
    }
#pragma unroll
    for (int r = 0; r < 8; ++r) {
      int m = r + half * 8;
      Ah1[w][m * AH1_S + nl]      = (_Float16)fmaxf(c0[r] + bE1_0, 0.f);
      Ah1[w][m * AH1_S + 16 + nl] = (_Float16)fmaxf(c1[r] + bE1_1, 0.f);
    }

    // ---- stage 2: em = h1 @ We2 + be2  [16x6], 1 WMMA ----
    {
      v8f ce = {};
      v16h a = ldsAFrag(&Ah1[w][0], AH1_S, lane, 0);
      ce = wmma16(a, ldsBFrag(wsw, 6, lane), ce);
#pragma unroll
      for (int r = 0; r < 8; ++r) {
        int m = r + half * 8;
        float v = ce[r] + bE2;
        if (nl < 6) {
          An[w][m * AN_S + 32 + nl] = (_Float16)v;   // feat_n[:,32..37]
          int e = base + m;
          if (e < E) out_em[(size_t)e * 6 + nl] = v;
        }
      }
    }

    // ---- stage 3: h2 = relu(feat_n @ Wn1 + bn1)  [16x64], 8 WMMA ----
    v8f d0 = {}, d1 = {}, d2 = {}, d3 = {};
#pragma unroll
    for (int ki = 0; ki < 2; ++ki) {
      v16h a = ldsAFrag(&An[w][0], AN_S, lane, ki * 32);
      d0 = wmma16(a, ldsBFrag(wsw, 7 + ki * 4 + 0, lane), d0);
      d1 = wmma16(a, ldsBFrag(wsw, 7 + ki * 4 + 1, lane), d1);
      d2 = wmma16(a, ldsBFrag(wsw, 7 + ki * 4 + 2, lane), d2);
      d3 = wmma16(a, ldsBFrag(wsw, 7 + ki * 4 + 3, lane), d3);
    }
#pragma unroll
    for (int r = 0; r < 8; ++r) {
      int m = r + half * 8;
      Ah2[w][m * AH2_S + nl]      = (_Float16)fmaxf(d0[r] + bN1_0, 0.f);
      Ah2[w][m * AH2_S + 16 + nl] = (_Float16)fmaxf(d1[r] + bN1_1, 0.f);
      Ah2[w][m * AH2_S + 32 + nl] = (_Float16)fmaxf(d2[r] + bN1_2, 0.f);
      Ah2[w][m * AH2_S + 48 + nl] = (_Float16)fmaxf(d3[r] + bN1_3, 0.f);
    }

    // ---- stage 4: msg = h2 @ Wn2 + bn2  [16x32], 4 WMMA; atomic scatter ----
    v8f m0 = {}, m1 = {};
#pragma unroll
    for (int ki = 0; ki < 2; ++ki) {
      v16h a = ldsAFrag(&Ah2[w][0], AH2_S, lane, ki * 32);
      m0 = wmma16(a, ldsBFrag(wsw, 15 + ki * 2 + 0, lane), m0);
      m1 = wmma16(a, ldsBFrag(wsw, 15 + ki * 2 + 1, lane), m1);
    }
#pragma unroll
    for (int r = 0; r < 8; ++r) {
      int m = r + half * 8;
      int e = base + m;
      if (e < E) {
        int d = dstL[w][m];
        atomicAdd(out_nm + (size_t)d * 32 + nl,      m0[r] + bN2_0);
        atomicAdd(out_nm + (size_t)d * 32 + 16 + nl, m1[r] + bN2_1);
      }
    }
  }
}

extern "C" void kernel_launch(void* const* d_in, const int* in_sizes, int n_in,
                              void* d_out, int out_size, void* d_ws, size_t ws_size,
                              hipStream_t stream) {
  (void)n_in; (void)out_size; (void)d_ws; (void)ws_size;
  const float* x_node = (const float*)d_in[0];
  const float* x_edge = (const float*)d_in[1];
  const int*   src    = (const int*)d_in[2];
  const int*   dst    = (const int*)d_in[3];
  const int*   b_mask = (const int*)d_in[4];
  const float* Wn1 = (const float*)d_in[5];
  const float* bn1 = (const float*)d_in[6];
  const float* Wn2 = (const float*)d_in[7];
  const float* bn2 = (const float*)d_in[8];
  const float* We1 = (const float*)d_in[9];
  const float* be1 = (const float*)d_in[10];
  const float* We2 = (const float*)d_in[11];
  const float* be2 = (const float*)d_in[12];

  const int N = in_sizes[0] / 32;   // nodes
  const int E = in_sizes[2];        // edges

  float* out_nm = (float*)d_out;
  float* out_em = out_nm + (size_t)N * 32;

  // atomics accumulate into out_nm -> zero it every call (graph-capture safe)
  hipMemsetAsync(out_nm, 0, (size_t)N * 32 * sizeof(float), stream);

  const int numTiles = (E + 15) / 16;
  int blocks = 4096;
  const int maxBlocks = (numTiles + WAVES - 1) / WAVES;
  if (blocks > maxBlocks) blocks = maxBlocks;
  if (blocks < 1) blocks = 1;

  mpn_wmma_kernel<<<dim3(blocks), dim3(WAVES * 32), 0, stream>>>(
      x_node, x_edge, src, dst, b_mask,
      Wn1, bn1, Wn2, bn2, We1, be1, We2, be2,
      out_nm, out_em, E, numTiles);
}
